// POSModel_36739150250636
// MI455X (gfx1250) — matrix operand order repaired and checked by reading, not compiled, not run on hardware
//
#include <hip/hip_runtime.h>
#include <hip/hip_bf16.h>
#include <stdint.h>

// CDNA5 (gfx1250) wave32 WMMA types
typedef __attribute__((ext_vector_type(16))) _Float16 v16h;
typedef __attribute__((ext_vector_type(8)))  float    v8f;

#define S_LEN 4096
#define L_CH  16
#define WD    128
#define CD    32
#define NF    5
#define KW    5
#define H     32
#define IN_DIM 133       // WD + NF
#define KPAD   160       // IN_DIM padded to multiple of 32 for WMMA K-chunks
#define NTAGS  45
#define NPAD   48        // NTAGS padded to multiple of 16

__device__ __forceinline__ uint32_t pack2h(float a, float b) {
    union { uint32_t u; _Float16 h[2]; } p;
    p.h[0] = (_Float16)a; p.h[1] = (_Float16)b;
    return p.u;
}

__device__ __forceinline__ float fast_sigmoid(float x) {
    return __builtin_amdgcn_rcpf(1.f + __expf(-x));
}
__device__ __forceinline__ float fast_tanh(float x) {
    float e = __expf(2.f * x);
    return (e - 1.f) * __builtin_amdgcn_rcpf(e + 1.f);
}

// ---------------------------------------------------------------------------
// Kernel 0: pack weights into f16, K-padded buffers for WMMA.
//   wih16 [256][160]: rows 0..127 = w_ih_f, 128..255 = w_ih_b, K 133..159 = 0
//   outw16 [48][64] : rows 0..44 = out_w, rows 45..47 = 0
// ---------------------------------------------------------------------------
__global__ void pack_weights(const float* __restrict__ w_ih_f,
                             const float* __restrict__ w_ih_b,
                             const float* __restrict__ out_w,
                             _Float16* __restrict__ wih16,
                             _Float16* __restrict__ outw16) {
    int gid = blockIdx.x * 256 + threadIdx.x;
    if (gid < 256 * KPAD) {
        int r = gid / KPAD, k = gid % KPAD;
        float v = 0.f;
        if (k < IN_DIM) v = (r < 128) ? w_ih_f[r * IN_DIM + k]
                                      : w_ih_b[(r - 128) * IN_DIM + k];
        wih16[gid] = (_Float16)v;
    } else if (gid < 256 * KPAD + NPAD * 64) {
        int j = gid - 256 * KPAD;
        int o = j / 64, k = j % 64;
        outw16[j] = (o < NTAGS) ? (_Float16)out_w[o * 64 + k] : (_Float16)0.f;
    }
}

// ---------------------------------------------------------------------------
// Kernel 1: per-word representation.
//   rep16[s][0..127]   = word_emb[word_ids[s]]          (f16)
//   rep16[s][128..132] = max_l conv1d(char_emb rows)    (f16)
//   rep16[s][133..159] = 0                              (K padding)
// One 128-thread block per word; word-emb path vectorized (float4 -> b64).
// ---------------------------------------------------------------------------
__global__ __launch_bounds__(128) void rep_kernel(
        const int*   __restrict__ word_ids,
        const int*   __restrict__ char_ids,
        const float* __restrict__ word_emb,
        const float* __restrict__ char_emb,
        const float* __restrict__ conv_w,   // [NF][CD][KW]
        const float* __restrict__ conv_b,
        _Float16*    __restrict__ rep16) {
    __shared__ float ce[L_CH * CD];   // [16][32]
    __shared__ float cv[NF * L_CH];   // [5][16]
    const int t = threadIdx.x;
    const int s = blockIdx.x;

    // word embedding gather -> f16 (threads 0..31, float4 in, uint2 out)
    if (t < 32) {
        const int wid = word_ids[s];
        float4 f = *(const float4*)(word_emb + wid * WD + t * 4);
        uint2 o;
        o.x = pack2h(f.x, f.y);
        o.y = pack2h(f.z, f.w);
        *(uint2*)(rep16 + s * KPAD + t * 4) = o;
    }

    // char embedding gather into LDS (512 floats, coalesced per l-row)
    #pragma unroll
    for (int e = t; e < L_CH * CD; e += 128) {
        int l = e >> 5, c = e & 31;
        int cid = char_ids[s * L_CH + l];
        ce[e] = char_emb[cid * CD + c];
    }
    __syncthreads();

    // conv (same padding, pad=2): thread t<80 computes conv[f][l]
    if (t < NF * L_CH) {
        int f = t >> 4, l = t & 15;
        float acc = conv_b[f];
        #pragma unroll
        for (int k = 0; k < KW; ++k) {
            int ll = l + k - 2;
            if (ll >= 0 && ll < L_CH) {
                const float* wp = conv_w + f * CD * KW + k;
                #pragma unroll
                for (int c = 0; c < CD; ++c)
                    acc += ce[ll * CD + c] * wp[c * KW];
            }
        }
        cv[f * L_CH + l] = acc;
    }
    __syncthreads();

    // max over L per filter, and zero-fill K padding
    if (t < NF) {
        float m = cv[t * L_CH];
        #pragma unroll
        for (int l = 1; l < L_CH; ++l) m = fmaxf(m, cv[t * L_CH + l]);
        rep16[s * KPAD + WD + t] = (_Float16)m;
    } else if (t < 32) {
        rep16[s * KPAD + WD + t] = (_Float16)0.f;   // cols 133..159
    }
}

// ---------------------------------------------------------------------------
// Kernel 2: pre = rep @ w_ih^T + (b_ih + b_hh), both directions fused.
// One wave32 per 16x16 tile: grid (256 M-tiles, 16 N-tiles); N 0..7 fwd,
// 8..15 bwd. Fragments are contiguous per lane -> global_load_b128 x2 each.
// Output stored GATE-INTERLEAVED: pre[s][unit][gate] so the scan kernel can
// fetch one float4 per step.
// ---------------------------------------------------------------------------
__global__ __launch_bounds__(32) void pre_gemm(
        const _Float16* __restrict__ rep16,   // [4096][160]
        const _Float16* __restrict__ wih16,   // [256][160]
        const float* __restrict__ b_ih_f, const float* __restrict__ b_hh_f,
        const float* __restrict__ b_ih_b, const float* __restrict__ b_hh_b,
        float* __restrict__ pre) {            // [2][4096][32][4]
    const int lane = threadIdx.x;
    const int mt = blockIdx.x;     // 0..255
    const int nt = blockIdx.y;     // 0..15
    const int half = lane >> 4;    // lane-half selects K sub-block
    const int lm = lane & 15;

    const _Float16* ap = rep16 + (mt * 16 + lm) * KPAD;   // A row (M)
    const _Float16* bp = wih16 + (nt * 16 + lm) * KPAD;   // B col (N = gate row)
    const int ks = half * 8;

    v8f acc = {};
    #pragma unroll
    for (int kc = 0; kc < KPAD / 32; ++kc) {
        union { uint4 q[2]; v16h v; } A, B;
        const int ka = kc * 32 + ks;          // A: VGPR0-3 = K ks..ks+7
        A.q[0] = *(const uint4*)(ap + ka);          // 16B contiguous
        A.q[1] = *(const uint4*)(ap + ka + 16);     // VGPR4-7 = K 16+ks..
        const int kk = kc * 32 + half * 8;    // B: same K order as A (ks.., 16+ks..)
        B.q[0] = *(const uint4*)(bp + kk);
        B.q[1] = *(const uint4*)(bp + kk + 16);
        acc = __builtin_amdgcn_wmma_f32_16x16x32_f16(
                false, A.v, false, B.v, (short)0, acc, false, false);
    }

    const int dir = nt >> 3;
    const int jj = (nt & 7) * 16 + lm;     // gate row 0..127 within direction
    const int gate = jj >> 5, unit = jj & 31;
    const float bias = dir ? (b_ih_b[jj] + b_hh_b[jj])
                           : (b_ih_f[jj] + b_hh_f[jj]);
    float* pd = pre + (size_t)dir * S_LEN * 128;
    #pragma unroll
    for (int r = 0; r < 8; ++r) {          // D: VGPR r -> M = r + 8*half
        int m = mt * 16 + r + half * 8;
        pd[m * 128 + unit * 4 + gate] = acc[r] + bias;   // gate-interleaved
    }
}

// ---------------------------------------------------------------------------
// Kernel 3: the sequential LSTM scan — the true critical path.
// One wave32 per direction; lane j owns hidden unit j. Entire w_hh slice for
// this lane (32 x float4 = 128 VGPRs) is REGISTER-RESIDENT: staged through
// LDS gate-interleaved once, then the fully-unrolled matvec reads registers
// only. Per step: 1 prefetched global b128 (pre), 1 LDS store + 8 LDS b128
// broadcast reads (h), 128 FMAs, 5 fast activations, 1 b32 store. The `pre`
// load for step t+1 is issued before step t's math to hide L2 latency.
// Workgroup is one wave, so __syncthreads lowers to S_NOP per the CDNA5 ISA.
// ---------------------------------------------------------------------------
__global__ __launch_bounds__(32) void lstm_scan(
        const float* __restrict__ pre,      // [2][4096][32][4]
        const float* __restrict__ w_hh_f,   // [128][32] gate-major (i,f,g,o)
        const float* __restrict__ w_hh_b,
        float* __restrict__ hbuf) {         // [2][4096][32]
    __shared__ float wp[32 * 132];          // [unit][k][gate] + pad
    __shared__ float hs[32];
    const int lane = threadIdx.x;
    const int dir  = blockIdx.x;
    const float* W = dir ? w_hh_b : w_hh_f;

    // W flat index = g*1024 + j*32 + k  ->  wp[j*132 + k*4 + g]
    for (int idx = lane; idx < 128 * 32; idx += 32) {
        int g = idx >> 10, j = (idx >> 5) & 31, k = idx & 31;
        wp[j * 132 + k * 4 + g] = W[idx];
    }
    __syncthreads();

    // hoist this lane's recurrent weights into registers (128 VGPRs)
    const float* wj = wp + lane * 132;
    float4 wreg[H];
    #pragma unroll
    for (int k = 0; k < H; ++k) wreg[k] = *(const float4*)(wj + 4 * k);

    const float* pd = pre  + (size_t)dir * S_LEN * 128;
    float*       hd = hbuf + (size_t)dir * S_LEN * H;

    float h = 0.f, c = 0.f;
    // prefetch step 0
    int s0 = dir ? (S_LEN - 1) : 0;
    float4 g4 = *(const float4*)(pd + s0 * 128 + lane * 4);

    for (int t = 0; t < S_LEN; ++t) {
        const int s = dir ? (S_LEN - 1 - t) : t;   // backward runs time-reversed
        // prefetch next step's pre-activations (address independent of h,c)
        const int tn = (t + 1 < S_LEN) ? (t + 1) : t;
        const int sn = dir ? (S_LEN - 1 - tn) : tn;
        float4 g4n = *(const float4*)(pd + sn * 128 + lane * 4);

        float gi = g4.x, gf = g4.y, gg = g4.z, go = g4.w;
        hs[lane] = h;
        __syncthreads();
        #pragma unroll
        for (int k0 = 0; k0 < H; k0 += 4) {
            float4 hv = *(const float4*)(hs + k0);
            gi += hv.x * wreg[k0].x + hv.y * wreg[k0+1].x + hv.z * wreg[k0+2].x + hv.w * wreg[k0+3].x;
            gf += hv.x * wreg[k0].y + hv.y * wreg[k0+1].y + hv.z * wreg[k0+2].y + hv.w * wreg[k0+3].y;
            gg += hv.x * wreg[k0].z + hv.y * wreg[k0+1].z + hv.z * wreg[k0+2].z + hv.w * wreg[k0+3].z;
            go += hv.x * wreg[k0].w + hv.y * wreg[k0+1].w + hv.z * wreg[k0+2].w + hv.w * wreg[k0+3].w;
        }
        __syncthreads();
        float i_ = fast_sigmoid(gi);
        float f_ = fast_sigmoid(gf);
        float g_ = fast_tanh(gg);
        float o_ = fast_sigmoid(go);
        c = f_ * c + i_ * g_;
        h = o_ * fast_tanh(c);
        hd[s * H + lane] = h;    // already in output order for both directions
        g4 = g4n;
    }
}

// ---------------------------------------------------------------------------
// Kernel 4: logits = [hf|hb] @ out_w^T + out_b via WMMA (K=64 -> 2 chunks,
// chunk 0 = hf, chunk 1 = hb). A built from float4 loads + packed cvt;
// B fragments are b128 loads. Grid (256 M-tiles, 3 N-tiles).
// ---------------------------------------------------------------------------
__global__ __launch_bounds__(32) void out_gemm(
        const float*    __restrict__ hbuf,    // [2][4096][32]
        const _Float16* __restrict__ outw16,  // [48][64]
        const float*    __restrict__ out_b,
        float*          __restrict__ out) {   // [4096][45]
    const int lane = threadIdx.x;
    const int mt = blockIdx.x;   // 0..255
    const int nt = blockIdx.y;   // 0..2
    const int half = lane >> 4;
    const int lm = lane & 15;

    const int m = mt * 16 + lm;
    const int n = nt * 16 + lm;
    const int ks = half * 8;
    const _Float16* bp = outw16 + n * 64;

    v8f acc = {};
    #pragma unroll
    for (int kc = 0; kc < 2; ++kc) {
        // A: this K-chunk comes wholly from one direction's h buffer
        const float* base = hbuf + (size_t)kc * S_LEN * H + m * H;
        float4 f0 = *(const float4*)(base + ks);
        float4 f1 = *(const float4*)(base + ks + 4);
        float4 f2 = *(const float4*)(base + 16 + ks);
        float4 f3 = *(const float4*)(base + 20 + ks);
        union { uint32_t u[8]; v16h v; } A;
        A.u[0] = pack2h(f0.x, f0.y); A.u[1] = pack2h(f0.z, f0.w);
        A.u[2] = pack2h(f1.x, f1.y); A.u[3] = pack2h(f1.z, f1.w);
        A.u[4] = pack2h(f2.x, f2.y); A.u[5] = pack2h(f2.z, f2.w);
        A.u[6] = pack2h(f3.x, f3.y); A.u[7] = pack2h(f3.z, f3.w);

        union { uint4 q[2]; v16h v; } B;
        const int kk = kc * 32 + half * 8;
        B.q[0] = *(const uint4*)(bp + kk);
        B.q[1] = *(const uint4*)(bp + kk + 16);

        acc = __builtin_amdgcn_wmma_f32_16x16x32_f16(
                false, A.v, false, B.v, (short)0, acc, false, false);
    }

    if (n < NTAGS) {
        const float bo = out_b[n];
        #pragma unroll
        for (int r = 0; r < 8; ++r) {
            int mm = mt * 16 + r + half * 8;
            out[mm * NTAGS + n] = acc[r] + bo;
        }
    }
}

// ---------------------------------------------------------------------------
extern "C" void kernel_launch(void* const* d_in, const int* in_sizes, int n_in,
                              void* d_out, int out_size, void* d_ws, size_t ws_size,
                              hipStream_t stream) {
    const int*   word_ids = (const int*)  d_in[0];
    const int*   char_ids = (const int*)  d_in[1];
    const float* word_emb = (const float*)d_in[2];
    const float* char_emb = (const float*)d_in[3];
    const float* conv_w   = (const float*)d_in[4];
    const float* conv_b   = (const float*)d_in[5];
    const float* w_ih_f   = (const float*)d_in[6];
    const float* w_hh_f   = (const float*)d_in[7];
    const float* b_ih_f   = (const float*)d_in[8];
    const float* b_hh_f   = (const float*)d_in[9];
    const float* w_ih_b   = (const float*)d_in[10];
    const float* w_hh_b   = (const float*)d_in[11];
    const float* b_ih_b   = (const float*)d_in[12];
    const float* b_hh_b   = (const float*)d_in[13];
    const float* out_w    = (const float*)d_in[14];
    const float* out_b    = (const float*)d_in[15];
    float* out = (float*)d_out;

    // workspace carve-up (256B aligned)
    char* ws = (char*)d_ws;
    _Float16* rep16  = (_Float16*)(ws + 0);                        // 4096*160*2 = 1310720
    _Float16* wih16  = (_Float16*)(ws + 1310720);                  // 256*160*2  =   81920
    _Float16* outw16 = (_Float16*)(ws + 1392640);                  // 48*64*2    =    6144
    float*    pre    = (float*)   (ws + 1398784);                  // 2*4096*128*4 = 4194304
    float*    hbuf   = (float*)   (ws + 5593088);                  // 2*4096*32*4  = 1048576

    // 0) pack weights to f16 padded
    {
        int total = 256 * KPAD + NPAD * 64;
        pack_weights<<<(total + 255) / 256, 256, 0, stream>>>(
            w_ih_f, w_ih_b, out_w, wih16, outw16);
    }
    // 1) per-word representation (gathers + conv + maxpool)
    rep_kernel<<<S_LEN, 128, 0, stream>>>(
        word_ids, char_ids, word_emb, char_emb, conv_w, conv_b, rep16);
    // 2) input projection GEMM for both directions (WMMA, b128 fragments)
    pre_gemm<<<dim3(S_LEN / 16, 16), 32, 0, stream>>>(
        rep16, wih16, b_ih_f, b_hh_f, b_ih_b, b_hh_b, pre);
    // 3) sequential scan: one wave per direction, register-resident weights
    lstm_scan<<<2, 32, 0, stream>>>(pre, w_hh_f, w_hh_b, hbuf);
    // 4) output projection GEMM (WMMA)
    out_gemm<<<dim3(S_LEN / 16, 3), 32, 0, stream>>>(hbuf, outw16, out_b, out);
}
